// MambaBlock_35854386987594
// MI455X (gfx1250) — matrix operand order repaired and checked by reading, not compiled, not run on hardware
//
#include <hip/hip_runtime.h>
#include <hip/hip_bf16.h>
#include <stdint.h>

// ---------------- problem constants ----------------
#define B_   4
#define S_   2048
#define D_   1024
#define DI_  2048
#define DS_  16
#define DC_  4
#define R_   (B_ * S_)          // 8192 token rows

// ---------------- WMMA vector types ----------------
typedef __attribute__((ext_vector_type(16))) __bf16 v16bf;
typedef __attribute__((ext_vector_type(8)))  float  v8f;
typedef __attribute__((ext_vector_type(4)))  int    v4i_t;

#ifndef __has_builtin
#define __has_builtin(x) 0
#endif

// ---- CDNA5 async global->LDS copy (ASYNCcnt-tracked), signature probe-verified ----
#if __has_builtin(__builtin_amdgcn_global_load_async_to_lds_b128)
#define HAVE_ASYNC 1
typedef __attribute__((address_space(1))) v4i_t as1_v4i;
typedef __attribute__((address_space(3))) v4i_t as3_v4i;
#define ASYNC_CP_B128(gp, lp) \
    __builtin_amdgcn_global_load_async_to_lds_b128( \
        (as1_v4i*)(void*)(gp), (as3_v4i*)(void*)(lp), 0, 0)
__device__ __forceinline__ void wait_async0() {
#if __has_builtin(__builtin_amdgcn_s_wait_asynccnt)
    __builtin_amdgcn_s_wait_asynccnt(0);
#else
    asm volatile("s_wait_asynccnt 0x0" ::: "memory");
#endif
}
#else
#define HAVE_ASYNC 0
#endif

// ---------------- helpers ----------------
__device__ __forceinline__ uint16_t f2bf(float f) {
    uint32_t u = __float_as_uint(f);
    return (uint16_t)((u + 0x7FFFu + ((u >> 16) & 1u)) >> 16);   // RNE
}
__device__ __forceinline__ float bf2f(uint16_t h) {
    return __uint_as_float(((uint32_t)h) << 16);
}
__device__ __forceinline__ float silu_f(float v) {
    return v / (1.0f + __expf(-v));
}

// ---------------- 1a) fp32 -> bf16 convert (activations) ----------------
__global__ void cvt_f32_bf16(const float* __restrict__ in, uint16_t* __restrict__ out, int n) {
    int i = blockIdx.x * blockDim.x + threadIdx.x;
    if (i < n) out[i] = f2bf(in[i]);
}

// ---------------- 1b) fp32 [K][N] -> bf16 [N][K] transpose (weights, one-time) ----------------
__global__ __launch_bounds__(256)
void transpose_f32_bf16(const float* __restrict__ in, uint16_t* __restrict__ out,
                        int K, int N)
{
    __shared__ uint16_t t[32][33];
    int n0 = blockIdx.x * 32, k0 = blockIdx.y * 32;
    int tx = threadIdx.x & 31, ty = threadIdx.x >> 5;   // ty: 0..7
    #pragma unroll
    for (int i = 0; i < 4; ++i) {
        int k = ty + i * 8;
        t[k][tx] = f2bf(in[(uint32_t)(k0 + k) * N + n0 + tx]);
    }
    __syncthreads();
    #pragma unroll
    for (int i = 0; i < 4; ++i) {
        int n = ty + i * 8;
        out[(uint32_t)(n0 + n) * K + k0 + tx] = t[tx][n];
    }
}

// ---------------- 2) bf16 WMMA GEMM: C[M,N] = A[M,K] * Bt[N,K]^T ----------------
// BM=128, BN=128, BK=32. 256 threads = 8 waves (4 M x 2 N); wave = 32x64 region
// = 2x4 v_wmma_f32_16x16x32_bf16. Both operands K-contiguous in global, so the
// whole tile stages via async global->LDS b128 (double-buffered); fragments are
// contiguous ds_load_b128 (k = 16*half + s, same order in A and B).
#define BM 128
#define BN 128
#define BK 32
#define ASTR 40   // padded LDS row stride (halfs); multiple of 8 -> 16B-aligned b128
#define BSTR 40
#define MT 2
#define NT 4

__global__ __launch_bounds__(256)
void gemm_bf16_wmma(const uint16_t* __restrict__ A,    // [M][K] bf16
                    const uint16_t* __restrict__ Bt,   // [N][K] bf16 (pre-transposed)
                    float* __restrict__ C, int M, int N, int K)
{
    __shared__ __align__(16) uint16_t Als[2][BM * ASTR];
    __shared__ __align__(16) uint16_t Bls[2][BN * BSTR];

    const int tid  = threadIdx.x;
    const int lane = tid & 31;
    const int wave = tid >> 5;
    const int wm   = wave & 3;     // 0..3  (M)
    const int wn   = wave >> 2;    // 0..1  (N)
    const int hi   = lane >> 4;    // lane half
    const int l16  = lane & 15;

    const uint32_t m0 = blockIdx.y * BM;
    const uint32_t n0 = blockIdx.x * BN;

    uint4 areg[2], breg[2];        // fallback staging regs (non-async path)

    // issue global reads for tile at k0 into LDS buffer `buf`
    auto issue_stage = [&](int k0, int buf) {
        #pragma unroll
        for (int r = 0; r < 2; ++r) {               // A & B: each 128x32 halfs = 512 b128
            int row  = (tid >> 1) + r * 128;
            int part = tid & 1;
            uint32_t aoff = (m0 + row) * (uint32_t)K + (uint32_t)k0 + part * 16;
            uint32_t boff = (n0 + row) * (uint32_t)K + (uint32_t)k0 + part * 16;
#if HAVE_ASYNC
            ASYNC_CP_B128(A  + aoff, &Als[buf][row * ASTR + part * 16]);
            ASYNC_CP_B128(Bt + boff, &Bls[buf][row * BSTR + part * 16]);
#else
            areg[r] = *reinterpret_cast<const uint4*>(A  + aoff);
            breg[r] = *reinterpret_cast<const uint4*>(Bt + boff);
#endif
        }
        (void)buf;
    };
    auto commit_stage = [&](int buf) {
#if !HAVE_ASYNC
        #pragma unroll
        for (int r = 0; r < 2; ++r) {
            int row  = (tid >> 1) + r * 128;
            int part = tid & 1;
            *reinterpret_cast<uint4*>(&Als[buf][row * ASTR + part * 16]) = areg[r];
            *reinterpret_cast<uint4*>(&Bls[buf][row * BSTR + part * 16]) = breg[r];
        }
#else
        (void)buf;
#endif
    };

    v8f acc[MT][NT] = {};

    issue_stage(0, 0);
    commit_stage(0);
#if HAVE_ASYNC
    wait_async0();
#endif
    __syncthreads();

    int buf = 0;
    for (int k0 = 0; k0 < K; k0 += BK) {
        const bool has_next = (k0 + BK) < K;
        if (has_next) issue_stage(k0 + BK, buf ^ 1);

        // ---- fragments ----
        union Frag { uint4 q[2]; v16bf v; };
        Frag afr[MT], bfr[NT];
        const uint16_t* Ab = Als[buf];
        const uint16_t* Bb = Bls[buf];
        #pragma unroll
        for (int mt = 0; mt < MT; ++mt) {
            int m = wm * 32 + mt * 16 + l16;
            afr[mt].q[0] = *reinterpret_cast<const uint4*>(&Ab[m * ASTR + 16 * hi]);
            afr[mt].q[1] = *reinterpret_cast<const uint4*>(&Ab[m * ASTR + 16 * hi + 8]);
        }
        #pragma unroll
        for (int nt = 0; nt < NT; ++nt) {
            int n = wn * 64 + nt * 16 + l16;
            const uint4* p = reinterpret_cast<const uint4*>(&Bb[n * BSTR + 16 * hi]);
            bfr[nt].q[0] = p[0];
            bfr[nt].q[1] = p[1];
        }

        #pragma unroll
        for (int mt = 0; mt < MT; ++mt)
            #pragma unroll
            for (int nt = 0; nt < NT; ++nt)
                acc[mt][nt] = __builtin_amdgcn_wmma_f32_16x16x32_bf16(
                    false, afr[mt].v, false, bfr[nt].v,
                    (short)0, acc[mt][nt], false, false);

        if (has_next) commit_stage(buf ^ 1);
#if HAVE_ASYNC
        wait_async0();
#endif
        __syncthreads();
        buf ^= 1;
    }

    // ---- store C (f32 layout: VGPR r -> M=r + 8*half, N=l16) ----
    #pragma unroll
    for (int mt = 0; mt < MT; ++mt) {
        #pragma unroll
        for (int nt = 0; nt < NT; ++nt) {
            uint32_t n     = n0 + wn * 64 + nt * 16 + l16;
            uint32_t mbase = m0 + wm * 32 + mt * 16 + 8 * hi;
            float* cp = C + mbase * (uint32_t)N + n;
            #pragma unroll
            for (int r = 0; r < 8; ++r)
                cp[(uint32_t)r * (uint32_t)N] = acc[mt][nt][r];
        }
    }
}

// ---------------- 3) causal depthwise conv1d + bias + SiLU -> bf16 ----------------
__global__ __launch_bounds__(256)
void conv_silu_kernel(const float* __restrict__ xz, const float* __restrict__ cw,
                      const float* __restrict__ cb, uint16_t* __restrict__ xcbf)
{
    uint32_t gid = blockIdx.x * 256 + threadIdx.x;    // over R_*DI_
    uint32_t d = gid & (DI_ - 1);
    uint32_t r = gid >> 11;                           // DI_ = 2048
    uint32_t b = r >> 11;                             // S_  = 2048
    int      s = (int)(r & (S_ - 1));
    float acc = cb[d];
    #pragma unroll
    for (int i = 0; i < DC_; ++i) {
        int ss = s - (DC_ - 1) + i;
        if (ss >= 0)
            acc += cw[d * DC_ + i] * xz[(b * S_ + (uint32_t)ss) * (2 * DI_) + d];
    }
    xcbf[r * DI_ + d] = f2bf(silu_f(acc));
}

// ---------------- 4) u[r][s] = xc[r,:] . w_B[:,s]  (K=2048, N=16) ----------------
__global__ __launch_bounds__(256)
void u_kernel(const uint16_t* __restrict__ xcbf, const float* __restrict__ wB,
              float* __restrict__ u)
{
    int st = threadIdx.x & 15;
    int rr = threadIdx.x >> 4;
    uint32_t r = blockIdx.x * 16 + rr;
    const uint16_t* xr = xcbf + r * DI_;
    float acc = 0.f;
    for (int d0 = 0; d0 < DI_; d0 += 8) {
        uint4 q = *reinterpret_cast<const uint4*>(xr + d0);
        const uint16_t* hh = reinterpret_cast<const uint16_t*>(&q);
        #pragma unroll
        for (int i = 0; i < 8; ++i)
            acc += bf2f(hh[i]) * wB[(d0 + i) * DS_ + st];
    }
    u[r * DS_ + st] = acc;
}

// ---------------- 5) chunked linear scan: h = 0.9 h + 0.1 u ----------------
#define NCH 16
#define CHL 128   // NCH * CHL == S_

__global__ void scan_partial(const float* __restrict__ u, float* __restrict__ carry)
{
    int gid = blockIdx.x * 256 + threadIdx.x;    // 0..1023
    int b  = gid >> 8;
    int c  = (gid >> 4) & 15;
    int st = gid & 15;
    float h = 0.f;
    uint32_t base = b * S_ + c * CHL;
    for (int i = 0; i < CHL; ++i)
        h = 0.9f * h + 0.1f * u[(base + i) * DS_ + st];
    carry[(b * DS_ + st) * NCH + c] = h;
}

__global__ void scan_carry(const float* __restrict__ carry, float* __restrict__ hin)
{
    int gid = threadIdx.x;                       // 0..63 = (b,st)
    if (gid >= B_ * DS_) return;
    float A = 1.f;
    for (int i = 0; i < CHL; ++i) A *= 0.9f;     // chunk decay 0.9^128
    float h = 0.f;
    for (int c = 0; c < NCH; ++c) {
        hin[gid * NCH + c] = h;                  // state entering chunk c
        h = A * h + carry[gid * NCH + c];
    }
}

__global__ void scan_final(const float* __restrict__ u, const float* __restrict__ hin,
                           float* __restrict__ hbuf)
{
    int gid = blockIdx.x * 256 + threadIdx.x;    // 0..1023
    int b  = gid >> 8;
    int c  = (gid >> 4) & 15;
    int st = gid & 15;
    float h = hin[(b * DS_ + st) * NCH + c];
    uint32_t base = b * S_ + c * CHL;
    for (int i = 0; i < CHL; ++i) {
        uint32_t r = base + i;
        h = 0.9f * h + 0.1f * u[r * DS_ + st];
        hbuf[r * DS_ + st] = h;
    }
}

// ---------------- 6) y = (h @ w_C) * silu(z) -> bf16 ----------------
__global__ __launch_bounds__(256)
void y_gate_kernel(const float* __restrict__ hbuf, const float* __restrict__ wC,
                   const float* __restrict__ xz, uint16_t* __restrict__ ybf)
{
    uint32_t gid = blockIdx.x * 256 + threadIdx.x;    // over R_*DI_
    uint32_t j = gid & (DI_ - 1);
    uint32_t r = gid >> 11;
    float acc = 0.f;
    #pragma unroll
    for (int s = 0; s < DS_; ++s)
        acc += hbuf[r * DS_ + s] * wC[s * DI_ + j];
    float z = xz[r * (2 * DI_) + DI_ + j];
    ybf[r * DI_ + j] = f2bf(acc * silu_f(z));
}

// ---------------- workspace layout (bytes) ----------------
static constexpr size_t OFF_XBF   = 0;                       // R_*D_ bf16      = 16 MB
static constexpr size_t OFF_WINT  = 16777216;                // [4096][1024] bf16 = 8 MB
static constexpr size_t OFF_WOUTT = 25165824;                // [1024][2048] bf16 = 4 MB
static constexpr size_t OFF_XZ    = 29360128;                // R_*2*DI_ f32    = 128 MB
static constexpr size_t OFF_XCBF  = 163577856;               // R_*DI_ bf16     = 32 MB
static constexpr size_t OFF_U     = 197132288;               // R_*DS_ f32      = 512 KB
static constexpr size_t OFF_H     = 197656576;               // R_*DS_ f32      = 512 KB
static constexpr size_t OFF_CARRY = 198180864;               // 1024 f32
static constexpr size_t OFF_HIN   = 198184960;               // 1024 f32
static constexpr size_t OFF_YBF   = 198189056;               // R_*DI_ bf16     = 32 MB

extern "C" void kernel_launch(void* const* d_in, const int* in_sizes, int n_in,
                              void* d_out, int out_size, void* d_ws, size_t ws_size,
                              hipStream_t stream)
{
    const float* x      = (const float*)d_in[0];   // [B,S,D]
    const float* w_in   = (const float*)d_in[1];   // [D, 2*DI]
    const float* conv_w = (const float*)d_in[2];   // [DI,1,DC]
    const float* conv_b = (const float*)d_in[3];   // [DI]
    const float* w_B    = (const float*)d_in[4];   // [DI,DS]
    const float* w_C    = (const float*)d_in[5];   // [DS,DI]
    const float* w_out  = (const float*)d_in[6];   // [DI,D]
    float* out = (float*)d_out;                    // [B,S,D]

    char* ws = (char*)d_ws;
    uint16_t* x_bf     = (uint16_t*)(ws + OFF_XBF);
    uint16_t* w_in_t   = (uint16_t*)(ws + OFF_WINT);   // [2*DI][D] bf16
    uint16_t* w_out_t  = (uint16_t*)(ws + OFF_WOUTT);  // [D][DI] bf16
    float*    xz       = (float*)   (ws + OFF_XZ);
    uint16_t* xc_bf    = (uint16_t*)(ws + OFF_XCBF);
    float*    u        = (float*)   (ws + OFF_U);
    float*    hbuf     = (float*)   (ws + OFF_H);
    float*    carry    = (float*)   (ws + OFF_CARRY);
    float*    hin      = (float*)   (ws + OFF_HIN);
    uint16_t* ybf      = (uint16_t*)(ws + OFF_YBF);

    // 1) precision staging: x -> bf16; weights -> transposed bf16 [N][K]
    cvt_f32_bf16<<<(R_ * D_) / 256, 256, 0, stream>>>(x, x_bf, R_ * D_);
    transpose_f32_bf16<<<dim3((2 * DI_) / 32, D_ / 32), 256, 0, stream>>>(
        w_in, w_in_t, D_, 2 * DI_);
    transpose_f32_bf16<<<dim3(D_ / 32, DI_ / 32), 256, 0, stream>>>(
        w_out, w_out_t, DI_, D_);

    // 2) in_proj: xz[8192,4096] = x[8192,1024] @ w_in[1024,4096]
    gemm_bf16_wmma<<<dim3((2 * DI_) / BN, R_ / BM), 256, 0, stream>>>(
        x_bf, w_in_t, xz, R_, 2 * DI_, D_);

    // 3) causal conv + SiLU
    conv_silu_kernel<<<(R_ * DI_) / 256, 256, 0, stream>>>(xz, conv_w, conv_b, xc_bf);

    // 4) u = xc @ w_B
    u_kernel<<<R_ / 16, 256, 0, stream>>>(xc_bf, w_B, u);

    // 5) parallel chunked scan
    scan_partial<<<4, 256, 0, stream>>>(u, carry);
    scan_carry<<<1, 64, 0, stream>>>(carry, hin);
    scan_final<<<4, 256, 0, stream>>>(u, hin, hbuf);

    // 6) y = (h @ w_C) * silu(z)
    y_gate_kernel<<<(R_ * DI_) / 256, 256, 0, stream>>>(hbuf, w_C, xz, ybf);

    // 7) out_proj: out[8192,1024] = y[8192,2048] @ w_out[2048,1024]
    gemm_bf16_wmma<<<dim3(D_ / BN, R_ / BM), 256, 0, stream>>>(
        ybf, w_out_t, out, R_, D_, DI_);
}
